// TritonModel_29583734735508
// MI455X (gfx1250) — matrix operand order repaired and checked by reading, not compiled, not run on hardware
//
#include <hip/hip_runtime.h>

// ---------------- problem constants ----------------
#define T_  8
#define B_  16
#define C_  64          // C_IN == C_OUT == 64
#define H_  64
#define W_  64
#define HW_ (H_ * W_)
#define KTAPS 9
#define INV_TAU 0.5f
#define V_TH 1.0f

// LDS layouts (ushort = raw bf16 bits)
#define CO_PER_WG   32                       // each workgroup handles half the output channels
#define WLDS_COSTR  (KTAPS * C_)             // 576 halves per co
#define SLAB_COLS   66                       // cols -1 .. 64 (halo)
#define SLAB_CISTR  C_                       // ci innermost, contiguous

typedef __bf16  bf16_t;
typedef bf16_t  v16bf __attribute__((ext_vector_type(16)));
typedef float   v8f   __attribute__((ext_vector_type(8)));

union FragU {
    uint4 q[2];
    v16bf v;
};

static __device__ inline unsigned short f32_to_bf16_rne(float f) {
    unsigned u = __float_as_uint(f);
    unsigned r = (u + 0x7FFFu + ((u >> 16) & 1u)) >> 16;
    return (unsigned short)r;
}

__global__ __launch_bounds__(256) void snn_conv_lif_kernel(
    const float* __restrict__ x,       // [T, B, C, H, W]
    const float* __restrict__ conv_w,  // [C_OUT, C_IN, 3, 3]
    float* __restrict__ out)           // [T, B, C, H, W] spikes
{
    __shared__ __align__(16) unsigned short wlds[CO_PER_WG * WLDS_COSTR];  // 36864 B
    __shared__ __align__(16) unsigned short slab[3 * SLAB_COLS * SLAB_CISTR]; // 25344 B

    const int tid    = threadIdx.x;
    const int h      = blockIdx.x;   // output row
    const int b      = blockIdx.y;   // batch image
    const int cohalf = blockIdx.z;   // 0 or 1 -> co 0..31 / 32..63

    // ---- stage weights into LDS as [co_local][tap][ci] bf16 ----
    for (int idx = tid; idx < CO_PER_WG * KTAPS * C_; idx += 256) {
        const int ci       = idx & (C_ - 1);
        const int rest     = idx >> 6;
        const int tap      = rest % KTAPS;
        const int co_local = rest / KTAPS;
        const int co       = cohalf * CO_PER_WG + co_local;
        const float wv = conv_w[(co * C_ + ci) * KTAPS + tap];
        wlds[co_local * WLDS_COSTR + tap * C_ + ci] = f32_to_bf16_rne(wv);
    }

    // ---- per-wave tile assignment ----
    const int lane    = tid & 31;
    const int wv      = tid >> 5;          // 0..7
    const int co_tile = wv >> 2;           // 0..1 (16 co each)
    const int sub     = wv & 3;            // 0..3 (16 cols each)
    const int hs      = lane >> 4;         // half-select
    const int mn      = lane & 15;         // M for A, N for B/D

    // A-fragment base pointers (per lane), fixed over t
    const unsigned short* a_base =
        &wlds[(co_tile * 16 + mn) * WLDS_COSTR + hs * 8];

    v8f vstate = {};   // LIF membrane potential, lives in registers across t

    for (int t = 0; t < T_; ++t) {
        __syncthreads();   // previous iteration's slab reads done (also covers weight fill)

        // ---- stage input slab x[t,b, :, h-1..h+1, -1..64] -> bf16 [row][col][ci] ----
        const float* xt = x + ((size_t)(t * B_ + b)) * C_ * HW_;
        for (int idx = tid; idx < 3 * SLAB_COLS * C_; idx += 256) {
            const int col = idx % SLAB_COLS;       // 0..65  (global w = col-1)
            const int r2  = idx / SLAB_COLS;
            const int ci  = r2 & (C_ - 1);
            const int row = r2 >> 6;               // 0..2   (global h = h+row-1)
            const int gh  = h + row - 1;
            const int gw  = col - 1;
            float v = 0.0f;
            if ((unsigned)gh < (unsigned)H_ && (unsigned)gw < (unsigned)W_)
                v = xt[(size_t)ci * HW_ + gh * W_ + gw];
            slab[(row * SLAB_COLS + col) * SLAB_CISTR + ci] = f32_to_bf16_rne(v);
        }
        __syncthreads();

        // ---- implicit GEMM: 9 taps x 2 ci-chunks of K=32 ----
        v8f acc = {};
        #pragma unroll
        for (int kh = 0; kh < 3; ++kh) {
            #pragma unroll
            for (int kw = 0; kw < 3; ++kw) {
                const int tap    = kh * 3 + kw;
                const int colidx = sub * 16 + mn + kw;   // 0..65
                const unsigned short* b_base =
                    &slab[(kh * SLAB_COLS + colidx) * SLAB_CISTR + hs * 8];
                #pragma unroll
                for (int chunk = 0; chunk < 2; ++chunk) {
                    // A: 16x32 bf16, lane holds K = hs*8+0..7 and 16+hs*8+0..7
                    const unsigned short* ap = a_base + tap * C_ + chunk * 32;
                    FragU ua;
                    ua.q[0] = *(const uint4*)(ap);        // 8 halves, K run 0
                    ua.q[1] = *(const uint4*)(ap + 16);   // 8 halves, K run 1
                    // B: 32x16 bf16, lane holds column mn, K = hs*8+0..7 and 16+hs*8+0..7
                    const unsigned short* bp = b_base + chunk * 32;
                    FragU ub;
                    ub.q[0] = *(const uint4*)(bp);
                    ub.q[1] = *(const uint4*)(bp + 16);
                    acc = __builtin_amdgcn_wmma_f32_16x16x32_bf16(
                        false, ua.v, false, ub.v, (short)0, acc, false, false);
                }
            }
        }

        // ---- fused LIF update + spike store ----
        // v += (z - v)/tau ; spike = v>=1 ; v = (1-spike)*v  (V_RESET = 0)
        vstate = vstate + (acc - vstate) * INV_TAU;

        const int wcol = sub * 16 + mn;
        float* outp = out + ((size_t)(t * B_ + b)) * C_ * HW_ + h * W_ + wcol;
        #pragma unroll
        for (int r = 0; r < 8; ++r) {
            const float vv = vstate[r];
            const float s  = (vv >= V_TH) ? 1.0f : 0.0f;
            vstate[r] = vv * (1.0f - s);
            const int co = cohalf * CO_PER_WG + co_tile * 16 + hs * 8 + r;
            outp[(size_t)co * HW_] = s;
        }
    }
}

extern "C" void kernel_launch(void* const* d_in, const int* in_sizes, int n_in,
                              void* d_out, int out_size, void* d_ws, size_t ws_size,
                              hipStream_t stream) {
    const float* x      = (const float*)d_in[0];
    const float* conv_w = (const float*)d_in[1];
    float*       out    = (float*)d_out;
    (void)in_sizes; (void)n_in; (void)out_size; (void)d_ws; (void)ws_size;

    dim3 grid(H_, B_, 2);   // (row, batch, co-half) -> 2048 workgroups
    dim3 block(256);        // 8 wave32s
    snn_conv_lif_kernel<<<grid, block, 0, stream>>>(x, conv_w, out);
}
